// StaticSparseAttention_19035295055958
// MI455X (gfx1250) — matrix operand order repaired and checked by reading, not compiled, not run on hardware
//
#include <hip/hip_runtime.h>
#include <hip/hip_bf16.h>

// ---------------------------------------------------------------------------
// Types for CDNA5 WMMA (wave32): v_wmma_f32_16x16x32_bf16
// ---------------------------------------------------------------------------
typedef __attribute__((ext_vector_type(8)))  __bf16   bf16x8;
typedef __attribute__((ext_vector_type(16))) __bf16   bf16x16;
typedef __attribute__((ext_vector_type(8)))  float    f32x8;
typedef __attribute__((ext_vector_type(4)))  uint32_t u32x4;
typedef __attribute__((ext_vector_type(8)))  uint32_t u32x8;

static constexpr int B_    = 2;
static constexpr int S_    = 4096;
static constexpr int H_    = 12;
static constexpr int D_    = 64;
static constexpr int E_    = H_ * D_;      // 768
static constexpr int CHUNK_ = 256;
static constexpr int NCH_  = S_ / CHUNK_;  // 16
static constexpr float SCALE_ = 0.125f;    // D^-0.5

// Build a 16-element bf16 A/B fragment from two 16-byte LDS reads.
__device__ __forceinline__ bf16x16 frag2(const __bf16* p0, const __bf16* p1) {
  bf16x8 lo = *(const bf16x8*)p0;
  bf16x8 hi = *(const bf16x8*)p1;
  return __builtin_shufflevector(lo, hi, 0,1,2,3,4,5,6,7,8,9,10,11,12,13,14,15);
}

__device__ __forceinline__ f32x8 wmma_bf16(bf16x16 a, bf16x16 b, f32x8 c) {
  // (neg_a, A, neg_b, B, c_mod, C, reuse_a, reuse_b)
  return __builtin_amdgcn_wmma_f32_16x16x32_bf16(false, a, false, b, (short)0, c,
                                                 false, false);
}

// Flat LDS pointer -> 32-bit LDS offset (LDS aperture keeps offset in addr[31:0]).
__device__ __forceinline__ uint32_t lds_off(const void* p) {
  return (uint32_t)(uintptr_t)p;
}

// GLOBAL_LOAD_ASYNC_TO_LDS_B128 (GVS mode): LDS[vdst] = MEM[saddr + voff]
// Tracked with ASYNCcnt; resolve with s_wait_asynccnt.
__device__ __forceinline__ void async_b128(uint32_t dst_lds, uint64_t sbase,
                                           uint32_t voff) {
  asm volatile("global_load_async_to_lds_b128 %0, %1, %2"
               :: "v"(dst_lds), "v"(voff), "s"(sbase) : "memory");
}
__device__ __forceinline__ void wait_async0() {
  asm volatile("s_wait_asynccnt 0" ::: "memory");
}

// Tensor Data Mover: 2-group descriptor form (tensors up to 2D).
__device__ __forceinline__ void tdm_load_to_lds(u32x4 g0, u32x8 g1) {
  asm volatile("tensor_load_to_lds %0, %1" :: "s"(g0), "s"(g1) : "memory");
}

// ---------------------------------------------------------------------------
// Elementwise converts
// ---------------------------------------------------------------------------
__global__ void k_f32_to_bf16(const float* __restrict__ src,
                              __bf16* __restrict__ dst, int n) {
  int i = blockIdx.x * blockDim.x + threadIdx.x;
  if (i < n) dst[i] = (__bf16)src[i];
}

// dst[n*K + k] = (bf16)src[k*N + n]   (weights go to [N][K] so B-fragments
// read 16 contiguous k values per lane -> ds_read_b128)
__global__ void k_transpose_bf16(const float* __restrict__ src,
                                 __bf16* __restrict__ dst, int K, int N) {
  int i = blockIdx.x * blockDim.x + threadIdx.x;
  if (i < K * N) {
    int n = i / K;
    int kk = i - n * K;
    dst[i] = (__bf16)src[kk * N + n];
  }
}

// ---------------------------------------------------------------------------
// Tiled bf16 WMMA GEMM: C[M,N] = A[M,K] * Bt[N,K]^T + bias
// Block tile 128x128, 256 threads = 8 waves (2x4), wave tile 64x32 (4x2 WMMA).
// A/B tiles staged with GLOBAL_LOAD_ASYNC_TO_LDS_B128 (no VGPR round-trip).
// MODE 0: scatter into q/k/v  [B,H,S,D] bf16.   MODE 1: write f32 to outf.
// ---------------------------------------------------------------------------
template<int MODE>
__global__ __launch_bounds__(256)
void k_gemm(const __bf16* __restrict__ A, const __bf16* __restrict__ Bt,
            const float* __restrict__ bias,
            __bf16* __restrict__ qb, __bf16* __restrict__ kb,
            __bf16* __restrict__ vb, float* __restrict__ outf, int K) {
  __shared__ __attribute__((aligned(16))) __bf16 As[128 * 32];
  __shared__ __attribute__((aligned(16))) __bf16 Bs[128 * 32];

  const int t    = threadIdx.x;
  const int lane = t & 31, wave = t >> 5;
  const int hi   = lane >> 4, ln = lane & 15;
  const int wm   = wave & 1, wn = wave >> 1;
  const int m0   = blockIdx.y * 128, n0 = blockIdx.x * 128;

  f32x8 acc[4][2];
  for (int a = 0; a < 4; a++)
    for (int b = 0; b < 2; b++)
      for (int e = 0; e < 8; e++) acc[a][b][e] = 0.f;

  const int srow = t >> 1, scol = (t & 1) * 16;
  const uint32_t ldsA = lds_off(&As[srow * 32 + scol]);
  const uint32_t ldsB = lds_off(&Bs[srow * 32 + scol]);
  const uint64_t baseA = (uint64_t)(uintptr_t)A;
  const uint64_t baseB = (uint64_t)(uintptr_t)Bt;
  const int nk = K / 32;
  for (int ko = 0; ko < nk; ++ko) {
    uint32_t aoff = ((uint32_t)(m0 + srow) * K + ko * 32 + scol) * 2u;
    uint32_t boff = ((uint32_t)(n0 + srow) * K + ko * 32 + scol) * 2u;
    async_b128(ldsA,      baseA, aoff);
    async_b128(ldsA + 16, baseA, aoff + 16);
    async_b128(ldsB,      baseB, boff);
    async_b128(ldsB + 16, baseB, boff + 16);
    if (ko + 1 < nk) {                       // global_prefetch_b8 of next tile
      __builtin_prefetch((const char*)A + aoff + 64, 0, 0);
      __builtin_prefetch((const char*)Bt + boff + 64, 0, 0);
    }
    wait_async0();
    __syncthreads();

    bf16x16 bfr[2];
    for (int nt = 0; nt < 2; nt++) {
      const __bf16* p = &Bs[(wn * 32 + nt * 16 + ln) * 32 + hi * 16];
      bfr[nt] = frag2(p, p + 8);
    }
    for (int mt = 0; mt < 4; mt++) {
      const __bf16* p = &As[(wm * 64 + mt * 16 + ln) * 32 + hi * 16];
      bf16x16 afr = frag2(p, p + 8);
      for (int nt = 0; nt < 2; nt++)
        acc[mt][nt] = wmma_bf16(afr, bfr[nt], acc[mt][nt]);
    }
    __syncthreads();
  }

  for (int mt = 0; mt < 4; mt++)
    for (int nt = 0; nt < 2; nt++)
      for (int e = 0; e < 8; e++) {
        int gm  = m0 + wm * 64 + mt * 16 + e + 8 * hi;
        int col = n0 + wn * 32 + nt * 16 + ln;
        float val = acc[mt][nt][e] + bias[col];
        if (MODE == 0) {
          int sel = col / E_;
          int rem = col - sel * E_;
          int h = rem >> 6, d = rem & 63;
          int b = gm >> 12, s = gm & (S_ - 1);
          size_t dst = (((size_t)(b * H_ + h)) * S_ + s) * D_ + d;
          __bf16 bv = (__bf16)val;
          if (sel == 0)      qb[dst] = bv;
          else if (sel == 1) kb[dst] = bv;
          else               vb[dst] = bv;
        } else {
          outf[(size_t)gm * E_ + col] = val;
        }
      }
}

// ---------------------------------------------------------------------------
// 2D RoPE, in place on q and k (skips CLS token s=0). One thread per pair.
// ---------------------------------------------------------------------------
__global__ void k_rope(__bf16* __restrict__ q, __bf16* __restrict__ kk,
                       const int* __restrict__ coords, int total) {
  int i = blockIdx.x * blockDim.x + threadIdx.x;
  if (i >= total) return;
  int p = i & 31;
  int r = i >> 5;
  int srel = r % (S_ - 1); r /= (S_ - 1);
  int h = r % H_;
  int b = r / H_;
  int s = srel + 1;
  float cx = (float)coords[((size_t)b * (S_ - 1) + srel) * 2 + 0];
  float cy = (float)coords[((size_t)b * (S_ - 1) + srel) * 2 + 1];
  int j = (p < 16) ? p : (p - 16);
  float inv = __expf(-(float)j * (9.2103403720f / 16.f));   // 10000^(-j/16)
  float xy  = ((p < 16) ? cx : cy) * 1e-5f;
  float sn, cs;
  __sincosf(xy * inv, &sn, &cs);
  size_t base = (((size_t)(b * H_ + h)) * S_ + s) * D_ + 2 * p;
  {
    float x0 = (float)q[base], x1 = (float)q[base + 1];
    q[base]     = (__bf16)(x0 * cs - x1 * sn);
    q[base + 1] = (__bf16)(x1 * cs + x0 * sn);
  }
  {
    float x0 = (float)kk[base], x1 = (float)kk[base + 1];
    kk[base]     = (__bf16)(x0 * cs - x1 * sn);
    kk[base + 1] = (__bf16)(x1 * cs + x0 * sn);
  }
}

// ---------------------------------------------------------------------------
// Windowed flash attention. 1 block = one (b,h,chunk). 256 threads = 8 waves,
// wave owns 32 query rows (2 WMMA m-tiles), thread t owns softmax of row t.
// Q chunk staged by the Tensor Data Mover; K tiles by async-to-LDS loads.
// Key stream: [CLS tile (1 valid key)] then 32-key tiles over the window.
// ---------------------------------------------------------------------------
__global__ __launch_bounds__(256)
void k_attn(const __bf16* __restrict__ q, const __bf16* __restrict__ k,
            const __bf16* __restrict__ v, __bf16* __restrict__ o) {
  __shared__ __attribute__((aligned(16))) __bf16 Qs[256 * 64];  // [row][d]
  __shared__ __attribute__((aligned(16))) __bf16 Ks[32 * 64];   // [key][d]
  __shared__ __attribute__((aligned(16))) __bf16 Vt[64 * 32];   // [d][key]
  __shared__ __attribute__((aligned(16))) float  Ps[256 * 32];  // raw scores
  __shared__ __attribute__((aligned(16))) __bf16 Pb[256 * 32];  // softmaxed P
  __shared__ float rowscale[256];

  const int t    = threadIdx.x;
  const int lane = t & 31, wave = t >> 5;
  const int hi   = lane >> 4, ln = lane & 15;
  const int blk  = blockIdx.x;
  const int c    = blk % NCH_;
  const int h    = (blk / NCH_) % H_;
  const int b    = blk / (NCH_ * H_);
  const size_t headbase = ((size_t)(b * H_ + h)) * S_ * D_;

  // Stage the 256x64 bf16 Q chunk (32 KB, contiguous) with the TDM.
  if (wave == 0) {
    const uint64_t ga = (uint64_t)(uintptr_t)(q + headbase +
                                              (size_t)(c * CHUNK_) * D_);
    const uint32_t la  = lds_off(&Qs[0]);
    const uint32_t NEL = 256 * 64;            // elements (data_size = 2B)
    u32x4 g0;
    g0[0] = 1u;                               // count=1 valid user descriptor
    g0[1] = la;                               // lds_addr
    g0[2] = (uint32_t)ga;                     // global_addr[31:0]
    g0[3] = (uint32_t)(ga >> 32) | (2u << 30);// global_addr[56:32] | type=2
    u32x8 g1;
    g1[0] = (1u << 16);                       // data_size=1 (2 bytes)
    g1[1] = (NEL & 0xffffu) << 16;            // tensor_dim0[15:0]
    g1[2] = (NEL >> 16) & 0xffffu;            // tensor_dim0[31:16], dim1=0... 
    g1[2] |= (1u << 16);                      // tensor_dim1 = 1
    g1[3] = (NEL & 0xffffu) << 16;            // tile_dim0 = 16384 (1-D tile)
    g1[4] = 0;                                // tile_dim1/2 unused
    g1[5] = NEL;                              // tensor_dim0_stride[31:0]
    g1[6] = 0;
    g1[7] = 0;
    tdm_load_to_lds(g0, g1);
    __builtin_amdgcn_s_wait_tensorcnt(0);     // TENSORcnt -> 0
  }

  const int row0 = wave * 32;
  float mi = -1e30f, li = 0.f;
  f32x8 Oacc[2][4];
  for (int a = 0; a < 2; a++)
    for (int n = 0; n < 4; n++)
      for (int e = 0; e < 8; e++) Oacc[a][n][e] = 0.f;

  const int range_lo = (c > 0 ? (c - 1) : 0) * CHUNK_;
  const int range_hi = ((c + 2) < NCH_ ? (c + 2) : NCH_) * CHUNK_;
  const int ntiles   = 1 + (range_hi - range_lo) / 32;

  const int kj = t >> 3;        // staging: key row 0..31
  const int kd = (t & 7) * 8;   // staging: d offset
  const uint64_t kbase = (uint64_t)(uintptr_t)k;
  const uint32_t ldsK  = lds_off(&Ks[kj * 64 + kd]);

  for (int it = 0; it < ntiles; ++it) {
    const bool clsTile = (it == 0);
    const int kb = clsTile ? 0 : range_lo + (it - 1) * 32;

    __syncthreads();  // previous AV + Q staging done before Ks/Vt overwrite
    {
      // K tile: async global->LDS (ASYNCcnt). V tile: VGPR path (transpose).
      uint32_t koff = (uint32_t)((headbase + (size_t)(kb + kj) * D_ + kd) * 2);
      async_b128(ldsK, kbase, koff);
      const __bf16* gv = v + headbase + (size_t)(kb + kj) * D_ + kd;
      bf16x8 vv = *(const bf16x8*)gv;
      for (int i = 0; i < 8; i++) Vt[(kd + i) * 32 + kj] = vv[i];  // transpose
      if (it + 1 < ntiles) {
        int nkb = clsTile ? range_lo : kb + 32;
        __builtin_prefetch(k + headbase + (size_t)(nkb + kj) * D_ + kd, 0, 0);
        __builtin_prefetch(v + headbase + (size_t)(nkb + kj) * D_ + kd, 0, 0);
      }
      wait_async0();
    }
    __syncthreads();

    // scores: S = Q (16x32,k-slices) x K^T (32x16) per 16x16 tile
    for (int mt = 0; mt < 2; ++mt) {
      const __bf16* pa = &Qs[(row0 + mt * 16 + ln) * 64 + hi * 16];
      bf16x16 a0 = frag2(pa,      pa + 8);        // d 0..31
      bf16x16 a1 = frag2(pa + 32, pa + 40);       // d 32..63
      for (int nt = 0; nt < 2; ++nt) {
        f32x8 sacc;
        for (int e = 0; e < 8; e++) sacc[e] = 0.f;
        const __bf16* pb = &Ks[(nt * 16 + ln) * 64 + hi * 16];
        sacc = wmma_bf16(a0, frag2(pb,      pb + 8),  sacc);
        sacc = wmma_bf16(a1, frag2(pb + 32, pb + 40), sacc);
        for (int e = 0; e < 8; e++)
          Ps[(row0 + mt * 16 + e + 8 * hi) * 32 + nt * 16 + ln] = sacc[e];
      }
    }
    __syncthreads();

    // online softmax: thread t owns query row t
    {
      float sc[32];
      float tmax = -1e30f;
      for (int j = 0; j < 32; ++j) {
        float sv = Ps[t * 32 + j] * SCALE_;
        bool valid = clsTile ? (j == 0) : ((kb + j) >= 1);  // gidx >= NUM_CLS
        if (!valid) sv = -1e9f;
        sc[j] = sv;
        tmax = fmaxf(tmax, sv);
      }
      float newm = fmaxf(mi, tmax);
      float rs = __expf(mi - newm);
      float psum = 0.f;
      for (int j = 0; j < 32; ++j) {
        float pv = __expf(sc[j] - newm);
        psum += pv;
        Pb[t * 32 + j] = (__bf16)pv;
      }
      li = li * rs + psum;
      mi = newm;
      rowscale[t] = rs;
    }
    __syncthreads();

    // rescale O, then O += P (16x32) x V (32x16 per d-slice)
    for (int mt = 0; mt < 2; ++mt) {
      for (int e = 0; e < 8; e++) {
        float rs = rowscale[row0 + mt * 16 + e + 8 * hi];
        for (int nt = 0; nt < 4; nt++) Oacc[mt][nt][e] *= rs;
      }
      const __bf16* pp = &Pb[(row0 + mt * 16 + ln) * 32 + hi * 16];
      bf16x16 pf = frag2(pp, pp + 8);
      for (int nt = 0; nt < 4; ++nt) {
        const __bf16* pv = &Vt[(nt * 16 + ln) * 32 + hi * 16];
        Oacc[mt][nt] = wmma_bf16(pf, frag2(pv, pv + 8), Oacc[mt][nt]);
      }
    }
  }

  __syncthreads();
  rowscale[t] = 1.f / li;
  __syncthreads();
  for (int mt = 0; mt < 2; ++mt)
    for (int e = 0; e < 8; e++) {
      int mrow = row0 + mt * 16 + e + 8 * hi;
      float inv = rowscale[mrow];
      int s = c * CHUNK_ + mrow;
      for (int nt = 0; nt < 4; nt++) {
        int d = nt * 16 + ln;
        o[((size_t)(b * S_) + s) * E_ + h * 64 + d] =
            (__bf16)(Oacc[mt][nt][e] * inv);
      }
    }
}

// ---------------------------------------------------------------------------
// CLS query: full attention over all S keys. 1 block per (b,h). Tiny.
// ---------------------------------------------------------------------------
__global__ __launch_bounds__(256)
void k_cls(const __bf16* __restrict__ q, const __bf16* __restrict__ k,
           const __bf16* __restrict__ v, __bf16* __restrict__ o) {
  __shared__ float qsh[64];
  __shared__ float red[256];
  __shared__ float osh[64];
  const int t = threadIdx.x;
  const int h = blockIdx.x % H_;
  const int b = blockIdx.x / H_;
  const size_t headbase = ((size_t)(b * H_ + h)) * S_ * D_;
  if (t < 64) { qsh[t] = (float)q[headbase + t]; osh[t] = 0.f; }
  __syncthreads();

  float sc[16];
  float lmax = -1e30f;
  for (int i = 0; i < 16; i++) {
    int s = t + i * 256;
    const __bf16* kr = k + headbase + (size_t)s * D_;
    float dot = 0.f;
    for (int d = 0; d < 64; ++d) dot += qsh[d] * (float)kr[d];
    sc[i] = dot * SCALE_;
    lmax = fmaxf(lmax, sc[i]);
  }
  red[t] = lmax; __syncthreads();
  for (int off = 128; off > 0; off >>= 1) {
    if (t < off) red[t] = fmaxf(red[t], red[t + off]);
    __syncthreads();
  }
  float M = red[0]; __syncthreads();

  float p[16];
  float lsum = 0.f;
  for (int i = 0; i < 16; i++) { p[i] = __expf(sc[i] - M); lsum += p[i]; }
  red[t] = lsum; __syncthreads();
  for (int off = 128; off > 0; off >>= 1) {
    if (t < off) red[t] += red[t + off];
    __syncthreads();
  }
  float SUM = red[0];

  float ac[64];
  for (int d = 0; d < 64; d++) ac[d] = 0.f;
  for (int i = 0; i < 16; i++) {
    int s = t + i * 256;
    const __bf16* vr = v + headbase + (size_t)s * D_;
    for (int d = 0; d < 64; d++) ac[d] += p[i] * (float)vr[d];
  }
  for (int d = 0; d < 64; d++) atomicAdd(&osh[d], ac[d]);
  __syncthreads();
  if (t < 64)
    o[((size_t)(b * S_)) * E_ + h * 64 + t] = (__bf16)(osh[t] / SUM);
}

// ---------------------------------------------------------------------------
extern "C" void kernel_launch(void* const* d_in, const int* in_sizes, int n_in,
                              void* d_out, int out_size, void* d_ws,
                              size_t ws_size, hipStream_t stream) {
  const float* x      = (const float*)d_in[0];
  const int*   coords = (const int*)d_in[1];
  const float* W_qkv  = (const float*)d_in[2];
  const float* b_qkv  = (const float*)d_in[3];
  const float* W_out  = (const float*)d_in[4];
  const float* b_out  = (const float*)d_in[5];
  float* out = (float*)d_out;

  char* ws = (char*)d_ws;
  size_t off = 0;
  auto alloc = [&](size_t bytes) {
    char* p = ws + off;
    off = (off + bytes + 255) & ~(size_t)255;
    return p;
  };
  const size_t nX = (size_t)B_ * S_ * E_;       // 6,291,456
  __bf16* xb  = (__bf16*)alloc(nX * 2);
  __bf16* qb  = (__bf16*)alloc(nX * 2);
  __bf16* kb  = (__bf16*)alloc(nX * 2);
  __bf16* vb  = (__bf16*)alloc(nX * 2);
  __bf16* ob  = (__bf16*)alloc(nX * 2);
  __bf16* wqT = (__bf16*)alloc((size_t)3 * E_ * E_ * 2);
  __bf16* woT = (__bf16*)alloc((size_t)E_ * E_ * 2);
  (void)in_sizes; (void)n_in; (void)out_size; (void)ws_size;

  k_f32_to_bf16<<<(int)((nX + 255) / 256), 256, 0, stream>>>(x, xb, (int)nX);
  k_transpose_bf16<<<(3 * E_ * E_ + 255) / 256, 256, 0, stream>>>(W_qkv, wqT,
                                                                  E_, 3 * E_);
  k_transpose_bf16<<<(E_ * E_ + 255) / 256, 256, 0, stream>>>(W_out, woT,
                                                              E_, E_);
  k_gemm<0><<<dim3(3 * E_ / 128, B_ * S_ / 128), 256, 0, stream>>>(
      xb, wqT, b_qkv, qb, kb, vb, nullptr, E_);
  {
    int total = B_ * H_ * (S_ - 1) * 32;
    k_rope<<<(total + 255) / 256, 256, 0, stream>>>(qb, kb, coords, total);
  }
  k_attn<<<B_ * H_ * NCH_, 256, 0, stream>>>(qb, kb, vb, ob);
  k_cls<<<B_ * H_, 256, 0, stream>>>(qb, kb, vb, ob);
  k_gemm<1><<<dim3(E_ / 128, B_ * S_ / 128), 256, 0, stream>>>(
      ob, woT, b_out, nullptr, nullptr, nullptr, out, E_);
}
